// Model_24180665876777
// MI455X (gfx1250) — matrix operand order repaired and checked by reading, not compiled, not run on hardware
//
#include <hip/hip_runtime.h>
#include <math.h>

// ---------------- model constants ----------------
#define B_      8
#define L_      512
#define N_      128
#define PRED_   96
#define DM_     512
#define H_      8
#define DFF_    2048
#define DK_     64
#define PATCH_  16
#define STRIDE_ 8
#define NPATCH_ 63
#define TMODES_ 31
#define VMODES_ 32
#define MROW_   (B_ * N_ * NPATCH_)     // 64512 == (B*N)*NPATCH == (B*NPATCH)*N
#define HEADK_  (NPATCH_ * DM_)         // 32256

typedef __bf16 v16bf __attribute__((ext_vector_type(16)));
typedef float  v8f   __attribute__((ext_vector_type(8)));

union AFrag { v16bf v; unsigned int u[8]; };
union BFrag { v16bf v; uint4 q[2]; };

__device__ __forceinline__ unsigned short f2bf(float f) {
    unsigned int u = __float_as_uint(f);
    u = (u + 0x7FFFu + ((u >> 16) & 1u)) >> 16;
    return (unsigned short)u;
}

// ---------------- f32 -> bf16 conversion ----------------
__global__ void cvt_bf16_kernel(const float* __restrict__ in,
                                unsigned short* __restrict__ out, long n) {
    long i = (long)blockIdx.x * blockDim.x + threadIdx.x;
    if (i < n) out[i] = f2bf(in[i]);
}

// ---------------- generic bf16 WMMA GEMM with async-LDS B staging -----------
// C(M,Nc) = A(M,K) @ W(Nc,K)^T [+bias] [+resid]
// EPI 0: f32 out (bias/resid optional).  EPI 1: exact GELU -> bf16 out.
// Block: 256 threads = 8 waves; block tile 128 rows x 64 cols; wave 16x64.
// B tile (64 cols x 32 K, shared by all 8 waves) is staged in LDS via
// global_load_async_to_lds_b128, double-buffered. K must be a multiple of 64.
#define BSTRIDE 40                    // padded K-stride (shorts): bank-conflict-free
#define BSTAGE  (64 * BSTRIDE)        // shorts per stage (5120 bytes)

template <int EPI>
__global__ __launch_bounds__(256) void gemm_bf16_kernel(
    const unsigned short* __restrict__ A,
    const unsigned short* __restrict__ W,
    const float* __restrict__ bias,
    const float* __restrict__ resid,
    float* __restrict__ Cf,
    unsigned short* __restrict__ Cb,
    int M, int Nc, int K)
{
    __shared__ unsigned short Bs[2 * BSTAGE];

    const int tid  = threadIdx.x;
    const int wave = tid >> 5;
    const int lane = tid & 31;
    const int half = lane >> 4;
    const int l15  = lane & 15;

    const int rowTile = (blockIdx.y * 8 + wave) * 16;  // M is a multiple of 128
    const int colBase = blockIdx.x * 64;

    // ---- async B staging: each thread owns one 16B chunk per stage ----
    const int bcol = tid >> 2;                 // local col 0..63
    const int bsub = tid & 3;                  // 8-short chunk within 32-K block
    int csrc = colBase + bcol;
    if (csrc >= Nc) csrc = Nc - 1;             // clamp loads; stores guarded below
    const unsigned short* gB = W + (long)csrc * K + bsub * 8;
    const unsigned ldsBase = (unsigned)(unsigned long long)(void*)&Bs[0];
    const unsigned ldsDst  = ldsBase + (unsigned)(bcol * BSTRIDE + bsub * 8) * 2u;

    const unsigned short* Arow = A + (long)(rowTile + l15) * K;

    bool okc[4];
#pragma unroll
    for (int ct = 0; ct < 4; ++ct) okc[ct] = (colBase + ct * 16) < Nc;

    const v8f zero = {0.f,0.f,0.f,0.f,0.f,0.f,0.f,0.f};
    v8f acc[4]; acc[0]=zero; acc[1]=zero; acc[2]=zero; acc[3]=zero;

    // issue stage0 (K block 0)
    {
        unsigned long long g = (unsigned long long)gB;
        asm volatile("global_load_async_to_lds_b128 %0, %1, off"
                     :: "v"(ldsDst), "v"(g) : "memory");
    }

    AFrag a0, a1;
#pragma unroll
    for (int j = 0; j < 8; ++j) {              // A frag for k0 = 0
        int kk = 2 * j + 16 * half;
        a0.u[j] = *reinterpret_cast<const unsigned int*>(Arow + kk);
    }

    for (int k0 = 0; k0 < K; k0 += 64) {
        // ---- issue stage1 (k0+32); wait for stage0; publish ----
        {
            unsigned long long g = (unsigned long long)(gB + k0 + 32);
            unsigned l = ldsDst + (unsigned)(BSTAGE * 2);
            asm volatile("global_load_async_to_lds_b128 %0, %1, off"
                         :: "v"(l), "v"(g) : "memory");
        }
        asm volatile("s_wait_asynccnt 0x1" ::: "memory");
        __syncthreads();
        // prefetch A frag for k0+32 (covered by the 4 WMMAs below)
#pragma unroll
        for (int j = 0; j < 8; ++j) {
            int kk = k0 + 32 + 2 * j + 16 * half;
            a1.u[j] = *reinterpret_cast<const unsigned int*>(Arow + kk);
        }
        // compute on stage0: fetch all 4 B frags first, then 4 WMMAs
        {
            BFrag b[4];
#pragma unroll
            for (int ct = 0; ct < 4; ++ct) {
                const uint4* bp = reinterpret_cast<const uint4*>(
                    &Bs[(ct * 16 + l15) * BSTRIDE + half * 16]);
                b[ct].q[0] = bp[0]; b[ct].q[1] = bp[1];
            }
#pragma unroll
            for (int ct = 0; ct < 4; ++ct)
                acc[ct] = __builtin_amdgcn_wmma_f32_16x16x32_bf16(
                    false, a0.v, false, b[ct].v, (short)0, acc[ct], false, false);
        }
        __syncthreads();                       // stage0 free for overwrite

        // ---- issue stage0 (k0+64, clamped dummy on last iter); wait stage1 ----
        int kn = (k0 + 64 < K) ? (k0 + 64) : 0;
        {
            unsigned long long g = (unsigned long long)(gB + kn);
            asm volatile("global_load_async_to_lds_b128 %0, %1, off"
                         :: "v"(ldsDst), "v"(g) : "memory");
        }
        asm volatile("s_wait_asynccnt 0x1" ::: "memory");
        __syncthreads();
        // prefetch A frag for kn
#pragma unroll
        for (int j = 0; j < 8; ++j) {
            int kk = kn + 2 * j + 16 * half;
            a0.u[j] = *reinterpret_cast<const unsigned int*>(Arow + kk);
        }
        // compute on stage1: fetch all 4 B frags first, then 4 WMMAs
        {
            BFrag b[4];
#pragma unroll
            for (int ct = 0; ct < 4; ++ct) {
                const uint4* bp = reinterpret_cast<const uint4*>(
                    &Bs[BSTAGE + (ct * 16 + l15) * BSTRIDE + half * 16]);
                b[ct].q[0] = bp[0]; b[ct].q[1] = bp[1];
            }
#pragma unroll
            for (int ct = 0; ct < 4; ++ct)
                acc[ct] = __builtin_amdgcn_wmma_f32_16x16x32_bf16(
                    false, a1.v, false, b[ct].v, (short)0, acc[ct], false, false);
        }
        __syncthreads();                       // stage1 free for overwrite
    }

    // ---- epilogue ----
#pragma unroll
    for (int ct = 0; ct < 4; ++ct) {
        if (!okc[ct]) continue;
        int c = colBase + ct * 16 + l15;
        float bv = bias ? bias[c] : 0.f;
#pragma unroll
        for (int e = 0; e < 8; ++e) {
            long rr = rowTile + 8 * half + e;  // C layout: M = e + 8*half
            float v = acc[ct][e] + bv;
            if (EPI == 1) {
                v = 0.5f * v * (1.f + erff(v * 0.70710678118f));
                Cb[rr * (long)Nc + c] = f2bf(v);
            } else {
                if (resid) v += resid[rr * (long)Nc + c];
                Cf[rr * (long)Nc + c] = v;
            }
        }
    }
}

// ---------------- moving-average decomposition ----------------
__global__ void decomp_kernel(const float* __restrict__ in,
                              float* __restrict__ seasonal,
                              float* __restrict__ trend,
                              int S, int T, int D)
{
    long idx = (long)blockIdx.x * blockDim.x + threadIdx.x;
    long total = (long)S * T * D;
    if (idx >= total) return;
    int d = (int)(idx % D);
    long st = idx / D;
    int t = (int)(st % T);
    int s = (int)(st / T);
    const float* base = in + ((long)s * T) * D + d;
    float sum = 0.f;
#pragma unroll 5
    for (int w = -12; w <= 12; ++w) {
        int tw = t + w;
        tw = tw < 0 ? 0 : (tw >= T ? T - 1 : tw);
        sum += base[(long)tw * D];
    }
    float m = sum * (1.f / 25.f);
    if (trend)    trend[idx] = m;
    if (seasonal) seasonal[idx] = in[idx] - m;
}

// ---------------- trend projection: einsum('bln,pl->bpn') + b ----------------
__global__ void trend_proj_kernel(const float* __restrict__ trend,
                                  const float* __restrict__ tW,
                                  const float* __restrict__ tb,
                                  float* __restrict__ out)
{
    int idx = blockIdx.x * blockDim.x + threadIdx.x;
    if (idx >= B_ * PRED_ * N_) return;
    int n = idx % N_;
    int p = (idx / N_) % PRED_;
    int b = idx / (N_ * PRED_);
    float s = 0.f;
    const float* tp = trend + (long)b * L_ * N_ + n;
    const float* wp = tW + (long)p * L_;
    for (int l = 0; l < L_; ++l) s += tp[(long)l * N_] * wp[l];
    out[idx] = s + tb[p];
}

// ---------------- patch embedding ----------------
__global__ void patch_embed_kernel(const float* __restrict__ seasonal,
                                   const float* __restrict__ pW,
                                   const float* __restrict__ pb,
                                   float* __restrict__ enc)
{
    __shared__ float sv[PATCH_];
    int blk = blockIdx.x;
    int pt = blk % NPATCH_;
    int n  = (blk / NPATCH_) % N_;
    int b  = blk / (NPATCH_ * N_);
    if (threadIdx.x < PATCH_)
        sv[threadIdx.x] = seasonal[((long)b * L_ + pt * STRIDE_ + threadIdx.x) * N_ + n];
    __syncthreads();
    long obase = (((long)b * N_ + n) * NPATCH_ + pt) * DM_;
    for (int d = threadIdx.x; d < DM_; d += blockDim.x) {
        float acc = pb[d];
        const float* w = pW + (long)d * PATCH_;
#pragma unroll
        for (int k = 0; k < PATCH_; ++k) acc += sv[k] * w[k];
        enc[obase + d] = acc;
    }
}

// ---------------- Fourier block ----------------
__global__ void dft_kernel(const float* __restrict__ q, float2* __restrict__ xft,
                           int S, int T, int modes)
{
    long idx = (long)blockIdx.x * blockDim.x + threadIdx.x;
    long total = (long)S * H_ * DK_ * modes;
    if (idx >= total) return;
    int m = (int)(idx % modes);
    long r = idx / modes;
    int i = (int)(r % DK_); r /= DK_;
    int h = (int)(r % H_);
    int s = (int)(r / H_);
    const float* qp = q + ((long)s * T) * DM_ + h * DK_ + i;
    float w0 = -6.2831853071795864f * (float)m / (float)T;
    float re = 0.f, im = 0.f;
    for (int t = 0; t < T; ++t) {
        float sn, cs;
        __sincosf(w0 * (float)t, &sn, &cs);
        float v = qp[(long)t * DM_];
        re += v * cs;
        im += v * sn;
    }
    xft[idx] = make_float2(re, im);
}

__global__ void modemix_kernel(const float2* __restrict__ xft,
                               const float* __restrict__ wr,
                               const float* __restrict__ wi,
                               float2* __restrict__ oft,
                               int S, int modes)
{
    long idx = (long)blockIdx.x * blockDim.x + threadIdx.x;
    long total = (long)S * H_ * DK_ * modes;
    if (idx >= total) return;
    int m = (int)(idx % modes);
    long r = idx / modes;
    int o = (int)(r % DK_); r /= DK_;
    int h = (int)(r % H_);
    int s = (int)(r / H_);
    const float2* xp = xft + (((long)s * H_ + h) * DK_) * (long)modes + m;
    long wbase = (((long)h * DK_) * DK_ + o) * (long)modes + m;
    long wstep = (long)DK_ * modes;
    float re = 0.f, im = 0.f;
    for (int i = 0; i < DK_; ++i) {
        float2 x = xp[(long)i * modes];
        float a = wr[wbase + (long)i * wstep];
        float c = wi[wbase + (long)i * wstep];
        re += x.x * a - x.y * c;
        im += x.x * c + x.y * a;
    }
    oft[idx] = make_float2(re, im);
}

__global__ void idft_kernel(const float2* __restrict__ oft, float* __restrict__ out,
                            int S, int T, int modes)
{
    long idx = (long)blockIdx.x * blockDim.x + threadIdx.x;
    long total = (long)S * T * DM_;
    if (idx >= total) return;
    int d = (int)(idx % DM_);
    long r = idx / DM_;
    int t = (int)(r % T);
    int s = (int)(r / T);
    int h = d >> 6, o = d & 63;
    const float2* op = oft + (((long)s * H_ + h) * DK_ + o) * (long)modes;
    float acc = op[0].x;
    float w0 = 6.2831853071795864f * (float)t / (float)T;
    for (int m = 1; m < modes; ++m) {
        float sn, cs;
        __sincosf(w0 * (float)m, &sn, &cs);
        float2 c = op[m];
        acc += 2.f * (c.x * cs - c.y * sn);
    }
    out[idx] = acc / (float)T;
}

// ---------------- LayerNorm over DM ----------------
__global__ void layernorm_kernel(const float* __restrict__ x,
                                 const float* __restrict__ w,
                                 const float* __restrict__ bb,
                                 float* __restrict__ out, int rows)
{
    int rIdx = blockIdx.x * blockDim.x + threadIdx.x;
    if (rIdx >= rows) return;
    const float* xp = x + (long)rIdx * DM_;
    float mu = 0.f;
    for (int d = 0; d < DM_; ++d) mu += xp[d];
    mu *= (1.f / DM_);
    float var = 0.f;
    for (int d = 0; d < DM_; ++d) { float t = xp[d] - mu; var += t * t; }
    var *= (1.f / DM_);
    float inv = rsqrtf(var + 1e-5f);
    float* op = out + (long)rIdx * DM_;
    for (int d = 0; d < DM_; ++d) op[d] = (xp[d] - mu) * inv * w[d] + bb[d];
}

__global__ void sub_seq_mean_kernel(const float* __restrict__ in,
                                    float* __restrict__ out, int S, int T)
{
    long idx = (long)blockIdx.x * blockDim.x + threadIdx.x;
    long total = (long)S * DM_;
    if (idx >= total) return;
    int d = (int)(idx % DM_);
    int s = (int)(idx / DM_);
    const float* p = in + ((long)s * T) * DM_ + d;
    float mu = 0.f;
    for (int t = 0; t < T; ++t) mu += p[(long)t * DM_];
    mu /= (float)T;
    float* op = out + ((long)s * T) * DM_ + d;
    for (int t = 0; t < T; ++t) op[(long)t * DM_] = p[(long)t * DM_] - mu;
}

// ---------------- permutes / fusion / final ----------------
__global__ void build_vin_kernel(const float* __restrict__ enc, float* __restrict__ vin)
{
    long idx = (long)blockIdx.x * blockDim.x + threadIdx.x;
    long total = (long)B_ * N_ * NPATCH_ * DM_;
    if (idx >= total) return;
    int d = (int)(idx % DM_);
    long r = idx / DM_;
    int pt = (int)(r % NPATCH_); r /= NPATCH_;
    int n  = (int)(r % N_);
    int b  = (int)(r / N_);
    vin[(((long)b * NPATCH_ + pt) * N_ + n) * DM_ + d] = enc[idx];
}

__global__ void fuse_kernel(const float* __restrict__ tout,
                            const float* __restrict__ vout,
                            const float* __restrict__ fusion,
                            unsigned short* __restrict__ fusedBf)
{
    long idx = (long)blockIdx.x * blockDim.x + threadIdx.x;
    long total = (long)B_ * N_ * NPATCH_ * DM_;
    if (idx >= total) return;
    int d = (int)(idx % DM_);
    long r = idx / DM_;
    int pt = (int)(r % NPATCH_); r /= NPATCH_;
    int n  = (int)(r % N_);
    int b  = (int)(r / N_);
    float f = 1.f / (1.f + __expf(-fusion[0]));
    float t = tout[idx];
    float v = vout[(((long)b * NPATCH_ + pt) * N_ + n) * DM_ + d];
    fusedBf[idx] = f2bf(f * t + (1.f - f) * v);
}

__global__ void final_kernel(const float* __restrict__ seas,
                             const float* __restrict__ headB,
                             const float* __restrict__ trendOut,
                             float* __restrict__ out)
{
    int idx = blockIdx.x * blockDim.x + threadIdx.x;
    if (idx >= B_ * PRED_ * N_) return;
    int n = idx % N_;
    int p = (idx / N_) % PRED_;
    int b = idx / (N_ * PRED_);
    out[idx] = seas[((long)b * N_ + n) * PRED_ + p] + headB[p] + trendOut[idx];
}

// ---------------- host-side helpers ----------------
static inline char* ws_take(void* base, size_t& off, size_t bytes) {
    char* p = (char*)base + off;
    off += (bytes + 255) & ~(size_t)255;
    return p;
}

static inline void launch_cvt(const float* src, unsigned short* dst, long n, hipStream_t s) {
    cvt_bf16_kernel<<<dim3((unsigned)((n + 255) / 256)), dim3(256), 0, s>>>(src, dst, n);
}

struct LayerPtrs {
    const float *bq, *bo, *fwr, *fwi;
    const unsigned short *WqB, *WoB, *c1B, *c2B;
};

struct EncScratch {
    float *tmp1, *tmp2;
    float2 *oft;
    unsigned short *xb, *hid;
};

static void run_gemm(const unsigned short* A, const unsigned short* W,
                     const float* bias, const float* resid,
                     float* Cf, unsigned short* Cb,
                     int M, int Nc, int K, int epi, hipStream_t s)
{
    dim3 grid((unsigned)((Nc + 63) / 64), (unsigned)(M / 128));
    dim3 blk(256);
    if (epi == 1)
        gemm_bf16_kernel<1><<<grid, blk, 0, s>>>(A, W, bias, resid, Cf, Cb, M, Nc, K);
    else
        gemm_bf16_kernel<0><<<grid, blk, 0, s>>>(A, W, bias, resid, Cf, Cb, M, Nc, K);
}

static void run_encoder(float* x, int S, int T, int modes,
                        const LayerPtrs* lyr,
                        const float* lnw, const float* lnb,
                        const EncScratch& sc, hipStream_t s)
{
    const int Mr = S * T;                       // 64512 for both encoders
    const long nMD  = (long)Mr * DM_;
    const long nFt  = (long)S * H_ * DK_ * modes;
    const unsigned cvtB = (unsigned)((nMD + 255) / 256);
    const unsigned ftB  = (unsigned)((nFt + 255) / 256);

    for (int l = 0; l < 2; ++l) {
        launch_cvt(x, sc.xb, nMD, s);
        run_gemm(sc.xb, lyr[l].WqB, lyr[l].bq, nullptr, sc.tmp1, nullptr, Mr, DM_, DM_, 0, s);
        dft_kernel<<<ftB, 256, 0, s>>>(sc.tmp1, (float2*)sc.tmp2, S, T, modes);
        modemix_kernel<<<ftB, 256, 0, s>>>((float2*)sc.tmp2, lyr[l].fwr, lyr[l].fwi, sc.oft, S, modes);
        idft_kernel<<<cvtB, 256, 0, s>>>(sc.oft, sc.tmp2, S, T, modes);
        launch_cvt(sc.tmp2, sc.xb, nMD, s);
        run_gemm(sc.xb, lyr[l].WoB, lyr[l].bo, x, sc.tmp1, nullptr, Mr, DM_, DM_, 0, s);
        decomp_kernel<<<cvtB, 256, 0, s>>>(sc.tmp1, x, nullptr, S, T, DM_);
        launch_cvt(x, sc.xb, nMD, s);
        run_gemm(sc.xb, lyr[l].c1B, nullptr, nullptr, nullptr, sc.hid, Mr, DFF_, DM_, 1, s);
        run_gemm(sc.hid, lyr[l].c2B, nullptr, x, sc.tmp1, nullptr, Mr, DM_, DFF_, 0, s);
        decomp_kernel<<<cvtB, 256, 0, s>>>(sc.tmp1, x, nullptr, S, T, DM_);
    }
    layernorm_kernel<<<(Mr + 255) / 256, 256, 0, s>>>(x, lnw, lnb, sc.tmp1, Mr);
    long nSD = (long)S * DM_;
    sub_seq_mean_kernel<<<(unsigned)((nSD + 255) / 256), 256, 0, s>>>(sc.tmp1, x, S, T);
}

// ---------------- entry point ----------------
extern "C" void kernel_launch(void* const* d_in, const int* in_sizes, int n_in,
                              void* d_out, int out_size, void* d_ws, size_t ws_size,
                              hipStream_t stream)
{
    (void)in_sizes; (void)n_in; (void)out_size; (void)ws_size;
    const float* x_enc   = (const float*)d_in[0];
    const float* patch_W = (const float*)d_in[4];
    const float* patch_b = (const float*)d_in[5];
    const float* trend_W = (const float*)d_in[6];
    const float* trend_b = (const float*)d_in[7];
    const float* head_W  = (const float*)d_in[8];
    const float* head_b  = (const float*)d_in[9];
    const float* fusion  = (const float*)d_in[10];
    const float* t_ln_w = (const float*)d_in[43];
    const float* t_ln_b = (const float*)d_in[44];
    const float* v_ln_w = (const float*)d_in[45];
    const float* v_ln_b = (const float*)d_in[46];

    size_t off = 0;
    float* seasonal = (float*)ws_take(d_ws, off, (size_t)B_ * L_ * N_ * 4);
    float* trend    = (float*)ws_take(d_ws, off, (size_t)B_ * L_ * N_ * 4);
    float* trendOut = (float*)ws_take(d_ws, off, (size_t)B_ * PRED_ * N_ * 4);
    float* enc  = (float*)ws_take(d_ws, off, (size_t)MROW_ * DM_ * 4);   // becomes t_out
    float* xV   = (float*)ws_take(d_ws, off, (size_t)MROW_ * DM_ * 4);   // becomes v_out
    float* tmp1 = (float*)ws_take(d_ws, off, (size_t)MROW_ * DM_ * 4);
    float* tmp2 = (float*)ws_take(d_ws, off, (size_t)MROW_ * DM_ * 4);   // also xft region
    float2* oft = (float2*)ws_take(d_ws, off, (size_t)B_ * N_ * H_ * DK_ * VMODES_ * 8 * 2);
    unsigned short* xb   = (unsigned short*)ws_take(d_ws, off, (size_t)MROW_ * DM_ * 2);
    unsigned short* hid  = (unsigned short*)ws_take(d_ws, off, (size_t)MROW_ * DFF_ * 2);
    unsigned short* fused= (unsigned short*)ws_take(d_ws, off, (size_t)B_ * N_ * HEADK_ * 2);
    float* seas = (float*)ws_take(d_ws, off, (size_t)B_ * N_ * PRED_ * 4);
    unsigned short* headWB = (unsigned short*)ws_take(d_ws, off, (size_t)PRED_ * HEADK_ * 2);

    LayerPtrs lyrT[2], lyrV[2];
    LayerPtrs* tabs[2] = { lyrT, lyrV };
    for (int enc_i = 0; enc_i < 2; ++enc_i) {
        for (int l = 0; l < 2; ++l) {
            int base = 11 + (enc_i * 2 + l) * 8;   // Wq,bq,Wo,bo,fwr,fwi,c1,c2
            LayerPtrs& P = tabs[enc_i][l];
            P.bq  = (const float*)d_in[base + 1];
            P.bo  = (const float*)d_in[base + 3];
            P.fwr = (const float*)d_in[base + 4];
            P.fwi = (const float*)d_in[base + 5];
            unsigned short* wq = (unsigned short*)ws_take(d_ws, off, (size_t)DM_ * DM_ * 2);
            unsigned short* wo = (unsigned short*)ws_take(d_ws, off, (size_t)DM_ * DM_ * 2);
            unsigned short* c1 = (unsigned short*)ws_take(d_ws, off, (size_t)DFF_ * DM_ * 2);
            unsigned short* c2 = (unsigned short*)ws_take(d_ws, off, (size_t)DM_ * DFF_ * 2);
            launch_cvt((const float*)d_in[base + 0], wq, (long)DM_ * DM_, stream);
            launch_cvt((const float*)d_in[base + 2], wo, (long)DM_ * DM_, stream);
            launch_cvt((const float*)d_in[base + 6], c1, (long)DFF_ * DM_, stream);
            launch_cvt((const float*)d_in[base + 7], c2, (long)DM_ * DFF_, stream);
            P.WqB = wq; P.WoB = wo; P.c1B = c1; P.c2B = c2;
        }
    }
    launch_cvt(head_W, headWB, (long)PRED_ * HEADK_, stream);

    long nBLN = (long)B_ * L_ * N_;
    decomp_kernel<<<(unsigned)((nBLN + 255) / 256), 256, 0, stream>>>(
        x_enc, seasonal, trend, B_, L_, N_);

    trend_proj_kernel<<<(B_ * PRED_ * N_ + 255) / 256, 256, 0, stream>>>(
        trend, trend_W, trend_b, trendOut);

    patch_embed_kernel<<<B_ * N_ * NPATCH_, 256, 0, stream>>>(
        seasonal, patch_W, patch_b, enc);

    long nEnc = (long)MROW_ * DM_;
    build_vin_kernel<<<(unsigned)((nEnc + 255) / 256), 256, 0, stream>>>(enc, xV);

    EncScratch sc{ tmp1, tmp2, oft, xb, hid };
    run_encoder(enc, B_ * N_,      NPATCH_, TMODES_, lyrT, t_ln_w, t_ln_b, sc, stream);
    run_encoder(xV,  B_ * NPATCH_, N_,      VMODES_, lyrV, v_ln_w, v_ln_b, sc, stream);

    fuse_kernel<<<(unsigned)((nEnc + 255) / 256), 256, 0, stream>>>(enc, xV, fusion, fused);

    run_gemm(fused, headWB, nullptr, nullptr, seas, nullptr,
             B_ * N_, PRED_, HEADK_, 0, stream);

    final_kernel<<<(B_ * PRED_ * N_ + 255) / 256, 256, 0, stream>>>(
        seas, head_b, trendOut, (float*)d_out);
}
